// TransFuse_53695681135127
// MI455X (gfx1250) — matrix operand order repaired and checked by reading, not compiled, not run on hardware
//
#include <hip/hip_runtime.h>
#include <math.h>

// ---------------------------------------------------------------------------
// TransFuse forward, MI455X (gfx1250).
// bf16 v_wmma_f32_16x16x32 GEMMs, fp32 accumulate; W.T*adj fused on the fly;
// adj tiles streamed with async global->LDS copies (ASYNCcnt) double-buffered
// one K-tile ahead; A/W tiles register-prefetched one K-tile ahead.
// Roofline: 1.2 GB weight traffic / 23.3 TB/s ~= 51 us floor; bf16 WMMA keeps
// the 158 GFLOP at a comparable ~60 us (fp32 16x16x4 would be ~8x slower).
// ---------------------------------------------------------------------------

#if __has_builtin(__builtin_amdgcn_global_load_async_to_lds_b128)
#define USE_ASYNC_LDS 1
#else
#define USE_ASYNC_LDS 0
#endif

typedef __attribute__((ext_vector_type(16))) __bf16 v16bf;
typedef __attribute__((ext_vector_type(8)))  __bf16 v8bf;
typedef __attribute__((ext_vector_type(8)))  float  v8f;

// async builtin operand type, matched to hipcc's diagnostic:
//   param0: 'int __vector(4) __device__ *'  (AS1)
//   param1: LDS side (AS3)
typedef int v4i_b128 __attribute__((vector_size(16)));
typedef __attribute__((address_space(1))) v4i_b128 gv4i_b128;
typedef __attribute__((address_space(3))) v4i_b128 lv4i_b128;

#define MBLK 128    // batch rows per block (512/128 = 4 M-blocks)
#define NBLK 128    // output features per block
#define KTILE 32    // WMMA K depth (bf16)
#define A_LD 40     // LDS pitch (bf16), padded vs 32 to spread banks
#define W_LD 40

template<int OFF>
__device__ __forceinline__ void adj_copy_b128(const float* g, float* l)
{
#if USE_ASYNC_LDS
    __builtin_amdgcn_global_load_async_to_lds_b128(
        (gv4i_b128*)(float*)g, (lv4i_b128*)l, OFF, 0);
#else
    *reinterpret_cast<float4*>(reinterpret_cast<char*>(l) + OFF) =
        *reinterpret_cast<const float4*>(reinterpret_cast<const char*>(g) + OFF);
#endif
}

__device__ __forceinline__ void wait_adj_copies()
{
#if USE_ASYNC_LDS
    asm volatile("s_wait_asynccnt 0x0" ::: "memory");
#endif
}

// out[b, colOff+n] = relu( sum_k A[b,k] * (W[n,k] * adj[k,n]) + bias[n] )
template<bool MASKED>
__global__ __launch_bounds__(256)
void gemm_relu_bf16(const float* __restrict__ A, int lda,
                    const float* __restrict__ W,    // N x K row major
                    const float* __restrict__ adj,  // K x N row major (MASKED)
                    const float* __restrict__ bias, // N
                    float* __restrict__ out, int ldout, int colOff,
                    int K, int N)
{
    // double-buffered adj tile (fp32), A/W tiles as bf16
    __shared__ __align__(16) float  adjLDS[MASKED ? 2 * KTILE * NBLK : 1]; // 32 KB
    __shared__ __align__(16) __bf16 A_lds[MBLK * A_LD];                    // 10 KB
    __shared__ __align__(16) __bf16 W_lds[NBLK * W_LD];                    // 10 KB

    const int tid   = threadIdx.x;
    const int lane  = tid & 31;
    const int wave  = tid >> 5;     // 0..7
    const int mw    = wave & 3;     // M strips mw and mw+4 (16 rows each)
    const int nw    = wave >> 2;    // 0..1 -> 64-col strip
    const int nloc  = lane & 15;
    const int khalf = lane >> 4;

    const int blockM = blockIdx.x * MBLK;
    const int blockN = blockIdx.y * NBLK;

    // Columns beyond N are never written by the async copies: zero them once
    // (both buffers). Visible to stage-2 readers via the in-loop barriers.
    if (MASKED && blockN + NBLK > N) {
        for (int idx = tid; idx < 2 * KTILE * NBLK; idx += 256) {
            int c = idx & (NBLK - 1);
            if (blockN + c >= N) adjLDS[idx] = 0.f;
        }
    }

    float4 rA[4], rW[4];    // register-prefetched tiles (tile kb while computing kb-1)
    auto loadA = [&](int kb0) {
        #pragma unroll
        for (int i = 0; i < 4; ++i) {
            int v = tid + i * 256;          // 128x32 / 4 = 1024 float4
            int m = v >> 3;
            int c = (v & 7) << 2;
            int gk = kb0 + c;
            float4 val = make_float4(0.f, 0.f, 0.f, 0.f);
            if (gk < K)                     // K%4==0 -> whole vec in/out
                val = *reinterpret_cast<const float4*>(A + (size_t)(blockM + m) * lda + gk);
            rA[i] = val;
        }
    };
    auto loadW = [&](int kb0) {
        #pragma unroll
        for (int i = 0; i < 4; ++i) {
            int v = tid + i * 256;          // 128n x 32k / 4
            int n = v >> 3;
            int c = (v & 7) << 2;
            int gn = blockN + n;
            int gk = kb0 + c;
            float4 val = make_float4(0.f, 0.f, 0.f, 0.f);
            if (gn < N && gk < K)
                val = *reinterpret_cast<const float4*>(W + (size_t)gn * K + gk);
            rW[i] = val;
        }
    };
    // adj tile: lane -> k row (per-lane EXEC guard only at the K edge, still
    // issues -> ASYNCcnt deterministic); wave+i -> column group (uniform guard).
    auto issueAdj = [&](int kb0, float* buf) {
        if (!MASKED) return;
        const float* g = adj + (size_t)(kb0 + lane) * N + blockN + wave * 4;
        float* l = buf + lane * NBLK + wave * 4;
        if (kb0 + lane < K) {
            if (blockN + (wave +  0) * 4 + 4 <= N) adj_copy_b128<  0>(g, l);
            if (blockN + (wave +  8) * 4 + 4 <= N) adj_copy_b128<128>(g, l);
            if (blockN + (wave + 16) * 4 + 4 <= N) adj_copy_b128<256>(g, l);
            if (blockN + (wave + 24) * 4 + 4 <= N) adj_copy_b128<384>(g, l);
        }
    };

    issueAdj(0, adjLDS);        // adj(0) in flight during prologue loads
    loadA(0); loadW(0);

    v8f acc[2][4] = {{v8f{}, v8f{}, v8f{}, v8f{}}, {v8f{}, v8f{}, v8f{}, v8f{}}};

    const int nkt = (K + KTILE - 1) / KTILE;
    for (int kb = 0; kb < nkt; ++kb) {
        const int kb0 = kb * KTILE;
        float* bufc = adjLDS + (MASKED ? (kb & 1) * (KTILE * NBLK) : 0);
        __syncthreads();        // previous tile's LDS reads complete

        // A regs -> bf16 LDS
        #pragma unroll
        for (int i = 0; i < 4; ++i) {
            int v = tid + i * 256;
            int m = v >> 3;
            int c = (v & 7) << 2;
            __bf16* dst = &A_lds[m * A_LD + c];
            dst[0] = (__bf16)rA[i].x; dst[1] = (__bf16)rA[i].y;
            dst[2] = (__bf16)rA[i].z; dst[3] = (__bf16)rA[i].w;
        }

        if (MASKED) {
            wait_adj_copies();              // adj(kb) landed in bufc
            int kedge = K - kb0;            // final partial K-tile: stale rows -> 0
            if (kedge < KTILE) {
                for (int idx = tid; idx < (KTILE - kedge) * NBLK; idx += 256)
                    bufc[kedge * NBLK + idx] = 0.f;
            }
        }
        __syncthreads();

        // Weff = W.T * adj -> bf16 LDS, n-major for B-fragment reads
        #pragma unroll
        for (int i = 0; i < 4; ++i) {
            int v = tid + i * 256;
            int n = v >> 3;
            int c = (v & 7) << 2;
            __bf16* dst = &W_lds[n * W_LD + c];
            if (MASKED) {
                dst[0] = (__bf16)(rW[i].x * bufc[(c + 0) * NBLK + n]);
                dst[1] = (__bf16)(rW[i].y * bufc[(c + 1) * NBLK + n]);
                dst[2] = (__bf16)(rW[i].z * bufc[(c + 2) * NBLK + n]);
                dst[3] = (__bf16)(rW[i].w * bufc[(c + 3) * NBLK + n]);
            } else {
                dst[0] = (__bf16)rW[i].x; dst[1] = (__bf16)rW[i].y;
                dst[2] = (__bf16)rW[i].z; dst[3] = (__bf16)rW[i].w;
            }
        }
        __syncthreads();

        // prefetch tile kb+1 (globals + async adj) while WMMAs execute
        if (kb + 1 < nkt) {
            issueAdj(kb0 + KTILE, adjLDS + (MASKED ? ((kb + 1) & 1) * (KTILE * NBLK) : 0));
            loadA(kb0 + KTILE);
            loadW(kb0 + KTILE);
        }

        // ---- 8 x v_wmma_f32_16x16x32_bf16 per wave per K-tile ----
        // A frag (ISA 7.12.2): lanes 0-15 K{0..7,16..23}, lanes 16-31 K{8..15,24..31}
        const __bf16* ar0 = &A_lds[(mw * 16 + nloc) * A_LD];
        const __bf16* ar1 = &A_lds[((mw + 4) * 16 + nloc) * A_LD];
        v8bf a00 = *reinterpret_cast<const v8bf*>(ar0 + khalf * 8);
        v8bf a01 = *reinterpret_cast<const v8bf*>(ar0 + 16 + khalf * 8);
        v8bf a10 = *reinterpret_cast<const v8bf*>(ar1 + khalf * 8);
        v8bf a11 = *reinterpret_cast<const v8bf*>(ar1 + 16 + khalf * 8);
        v16bf af0 = __builtin_shufflevector(a00, a01,
            0,1,2,3,4,5,6,7,8,9,10,11,12,13,14,15);
        v16bf af1 = __builtin_shufflevector(a10, a11,
            0,1,2,3,4,5,6,7,8,9,10,11,12,13,14,15);

        #pragma unroll
        for (int s = 0; s < 4; ++s) {
            // B frag: lane = column n, K{khalf*8..+7} then K{16+khalf*8..+7} (matches A)
            const __bf16* br = &W_lds[(nw * 64 + s * 16 + nloc) * W_LD + khalf * 8];
            v8bf b0 = *reinterpret_cast<const v8bf*>(br);
            v8bf b1 = *reinterpret_cast<const v8bf*>(br + 16);
            v16bf bf = __builtin_shufflevector(b0, b1,
                0,1,2,3,4,5,6,7,8,9,10,11,12,13,14,15);
            acc[0][s] = __builtin_amdgcn_wmma_f32_16x16x32_bf16(
                false, af0, false, bf, (short)0, acc[0][s], false, false);
            acc[1][s] = __builtin_amdgcn_wmma_f32_16x16x32_bf16(
                false, af1, false, bf, (short)0, acc[1][s], false, false);
        }
    }

    // ---- epilogue: bias + ReLU; C layout: VGPR r -> row r + khalf*8 ----
    #pragma unroll
    for (int s = 0; s < 4; ++s) {
        int col = blockN + nw * 64 + s * 16 + nloc;
        if (col < N) {
            float bv = bias[col];
            #pragma unroll
            for (int t = 0; t < 2; ++t) {
                #pragma unroll
                for (int r = 0; r < 8; ++r) {
                    int row = blockM + (mw + t * 4) * 16 + khalf * 8 + r;
                    float vv = acc[t][s][r] + bv;
                    out[(size_t)row * ldout + colOff + col] = vv > 0.f ? vv : 0.f;
                }
            }
        }
    }
}

// y[b] = sigmoid( h2[b,:] . Wout + bout ), exact fp32 (512x256x1)
__global__ __launch_bounds__(256)
void head_sigmoid(const float* __restrict__ h2, const float* __restrict__ Wout,
                  const float* __restrict__ bout, float* __restrict__ y)
{
    int b = blockIdx.x * blockDim.x + threadIdx.x;
    if (b >= 512) return;
    const float* row = h2 + (size_t)b * 256;
    float s = bout[0];
    #pragma unroll 4
    for (int j = 0; j < 256; ++j) s += row[j] * Wout[j];
    y[b] = 1.f / (1.f + expf(-s));
}

extern "C" void kernel_launch(void* const* d_in, const int* in_sizes, int n_in,
                              void* d_out, int out_size, void* d_ws, size_t ws_size,
                              hipStream_t stream)
{
    const float* in_mat = (const float*)d_in[0];
    const float* adj_sg = (const float*)d_in[1];
    const float* adj_gp = (const float*)d_in[2];
    const float* adj_br = (const float*)d_in[3];
    const float* adj_pp = (const float*)d_in[4];
    const float* W_sg  = (const float*)d_in[5];   const float* b_sg  = (const float*)d_in[6];
    const float* W_gp  = (const float*)d_in[7];   const float* b_gp  = (const float*)d_in[8];
    const float* W_br  = (const float*)d_in[9];   const float* b_br  = (const float*)d_in[10];
    const float* W_pp  = (const float*)d_in[11];  const float* b_pp  = (const float*)d_in[12];
    const float* W_h1  = (const float*)d_in[13];  const float* b_h1  = (const float*)d_in[14];
    const float* W_h2  = (const float*)d_in[15];  const float* b_h2  = (const float*)d_in[16];
    const float* W_out = (const float*)d_in[17];  const float* b_out = (const float*)d_in[18];

    // workspace: concat buffers realized via shared row strides
    float* bridge = (float*)d_ws;                       // 512 x 8000 (out_snp | out_gene)
    float* fuse   = bridge + (size_t)512 * 8000;        // 512 x 6000 (out_bridge | out_protein)
    float* h1     = fuse   + (size_t)512 * 6000;        // 512 x 1024
    float* h2     = h1     + (size_t)512 * 1024;        // 512 x 256
    (void)ws_size; (void)in_sizes; (void)n_in; (void)out_size;

    dim3 blk(256);
    auto launch = [&](const float* Ax, int lda, const float* Wx, const float* adjx,
                      const float* bx, float* ox, int ldout, int colOff,
                      int K, int N) {
        dim3 grd(512 / MBLK, (N + NBLK - 1) / NBLK);   // M-major -> L2 W-slice sharing
        if (adjx)
            hipLaunchKernelGGL((gemm_relu_bf16<true>),  grd, blk, 0, stream,
                               Ax, lda, Wx, adjx, bx, ox, ldout, colOff, K, N);
        else
            hipLaunchKernelGGL((gemm_relu_bf16<false>), grd, blk, 0, stream,
                               Ax, lda, Wx, (const float*)nullptr, bx, ox, ldout, colOff, K, N);
    };

    launch(in_mat + 8000, 28000, W_sg, adj_sg, b_sg, bridge, 8000, 0,    20000, 5000);
    launch(in_mat + 3000, 28000, W_gp, adj_gp, b_gp, bridge, 8000, 5000, 5000,  3000);
    launch(bridge,        8000,  W_br, adj_br, b_br, fuse,   6000, 0,    8000,  3000);
    launch(in_mat,        28000, W_pp, adj_pp, b_pp, fuse,   6000, 3000, 3000,  3000);
    launch(fuse,          6000,  W_h1, nullptr, b_h1, h1,    1024, 0,    6000,  1024);
    launch(h1,            1024,  W_h2, nullptr, b_h2, h2,    256,  0,    1024,  256);
    head_sigmoid<<<dim3(2), blk, 0, stream>>>(h2, W_out, b_out, (float*)d_out);
}
